// SpikingPolicyNet_40750649705168
// MI455X (gfx1250) — compile-verified, matrix operand order checked by reading, not run on hardware
//
#include <hip/hip_runtime.h>
#include <hip/hip_bf16.h>
#include <stdint.h>

// ---------------------------------------------------------------------------
// SpikingPolicyNet on gfx1250: bf16 WMMA GEMMs with fused LIF epilogues.
// B=8192, D_in=256, H=1024, D_out=64, T=15, TAU=20, V_TH=1.
// Pipeline: double-buffered LDS fed by async global->LDS DMA (ASYNCcnt).
// ---------------------------------------------------------------------------

typedef __attribute__((ext_vector_type(16))) __bf16 v16bf;
typedef __attribute__((ext_vector_type(8)))  float  v8f;
typedef int v4i __attribute__((vector_size(16)));   // matches builtin param type

#define TAU_INV (1.0f / 20.0f)
#define V_TH    1.0f

#define ASG __attribute__((address_space(1)))
#define ASL __attribute__((address_space(3)))
typedef ASG v4i* gv4i_p;   // global-AS pointer to int4
typedef ASL v4i* lv4i_p;   // LDS-AS pointer to int4

#if defined(__has_builtin)
#if __has_builtin(__builtin_amdgcn_global_load_async_to_lds_b128)
#define USE_ASYNC_LDS 1
#endif
#endif
#ifndef USE_ASYNC_LDS
#define USE_ASYNC_LDS 0
#endif

__device__ __forceinline__ void wait_async0() {
#if __has_builtin(__builtin_amdgcn_s_wait_asynccnt)
    __builtin_amdgcn_s_wait_asynccnt(0);
#else
    asm volatile("s_wait_asynccnt 0x0" ::: "memory");
#endif
}

__device__ __forceinline__ unsigned short f32_to_bf16(float x) {
    union { float f; unsigned u; } c; c.f = x;
    unsigned u = c.u;
    u += 0x7FFFu + ((u >> 16) & 1u);   // round-to-nearest-even
    return (unsigned short)(u >> 16);
}

// f32 -> bf16 (raw bits) with scale
__global__ __launch_bounds__(256) void cvt_bf16_kernel(
    const float* __restrict__ in, unsigned short* __restrict__ out,
    float scale, int n)
{
    int i = blockIdx.x * blockDim.x + threadIdx.x;
    int stride = gridDim.x * blockDim.x;
    for (; i < n; i += stride) out[i] = f32_to_bf16(in[i] * scale);
}

// Layer-1 LIF: v1 += (I1 - v1)/TAU; s = (v1 > 1); v1 *= (1-s); emit s as bf16.
__global__ __launch_bounds__(256) void lif1_kernel(
    const float4* __restrict__ I1, float4* __restrict__ v1,
    unsigned short* __restrict__ s1, int n4)
{
    int i = blockIdx.x * blockDim.x + threadIdx.x;
    int stride = gridDim.x * blockDim.x;
    for (; i < n4; i += stride) {
        float4 I = I1[i];
        float4 v = v1[i];
        ushort4 s;
        float vv, sp;
        vv = v.x + (I.x - v.x) * TAU_INV; sp = (vv > V_TH) ? 1.0f : 0.0f;
        v.x = vv * (1.0f - sp); s.x = sp != 0.0f ? 0x3F80u : 0u;
        vv = v.y + (I.y - v.y) * TAU_INV; sp = (vv > V_TH) ? 1.0f : 0.0f;
        v.y = vv * (1.0f - sp); s.y = sp != 0.0f ? 0x3F80u : 0u;
        vv = v.z + (I.z - v.z) * TAU_INV; sp = (vv > V_TH) ? 1.0f : 0.0f;
        v.z = vv * (1.0f - sp); s.z = sp != 0.0f ? 0x3F80u : 0u;
        vv = v.w + (I.w - v.w) * TAU_INV; sp = (vv > V_TH) ? 1.0f : 0.0f;
        v.w = vv * (1.0f - sp); s.w = sp != 0.0f ? 0x3F80u : 0u;
        v1[i] = v;
        *reinterpret_cast<ushort4*>(s1 + (size_t)i * 4) = s;
    }
}

// ---------------------------------------------------------------------------
// C(MxN) = A(MxK,bf16) @ B(NxK,bf16)^T + bias.
// 128x64 tile / block, 8 waves, each wave: 2x2 grid of 16x16 WMMA tiles.
// Double-buffered LDS; async global->LDS DMA when the builtin exists.
// EPI==0: Out[m*N+n] = acc + bias[n]
// EPI==1: fused layer-2 LIF: RMW v2 (=out_or_v) and ssum.
// ---------------------------------------------------------------------------
constexpr int LDT   = 40;   // LDS row stride (ushorts): 80B, 16B-aligned, bank-spread
constexpr int BLK_M = 128;
constexpr int BLK_N = 64;

template <int EPI>
__global__ __launch_bounds__(256) void wmma_gemm_kernel(
    const unsigned short* __restrict__ A,
    const unsigned short* __restrict__ Bm,
    const float* __restrict__ bias,
    float* __restrict__ out_or_v,
    float* __restrict__ ssum,
    int M, int N, int K)
{
    __shared__ unsigned short sA[2][BLK_M * LDT];
    __shared__ unsigned short sB[2][BLK_N * LDT];

    const int m_base = blockIdx.y * BLK_M;
    const int n_base = blockIdx.x * BLK_N;
    const int t    = threadIdx.x;
    const int wave = t >> 5;
    const int lane = t & 31;

    // cooperative tile load: each thread owns three 16B chunks per K-step
    //   A: rows t>>2 and 64+(t>>2); B: row t>>2; col (t&3)*8
    const int ldRow = t >> 2;          // 0..63
    const int ldCol = (t & 3) * 8;     // 0,8,16,24
    const unsigned short* gA0 = A  + (size_t)(m_base + ldRow)      * K + ldCol;
    const unsigned short* gA1 = A  + (size_t)(m_base + 64 + ldRow) * K + ldCol;
    const unsigned short* gB  = Bm + (size_t)(n_base + ldRow)      * K + ldCol;
    const int dA0 = ldRow * LDT + ldCol;
    const int dA1 = (ldRow + 64) * LDT + ldCol;
    const int dB  = ldRow * LDT + ldCol;

    // wave -> 2 M-subtiles (64 apart) x 2 N-subtiles (32 apart)
    const int ma0 = (wave & 3) * 16;
    const int ma1 = ma0 + 64;
    const int nb0 = (wave >> 2) * 16;
    const int nb1 = nb0 + 32;

    // ISA bf16 A/B fragment layout: lanes 0-15 K={0..7,16..23}, 16-31 K={8..15,24..31}
    const int lrow  = lane & 15;
    const int khalf = (lane >> 4) * 8;

    v8f acc00 = {};
    v8f acc01 = {};
    v8f acc10 = {};
    v8f acc11 = {};

#if USE_ASYNC_LDS
    // ---- async DMA double-buffer pipeline ----
    __builtin_amdgcn_global_load_async_to_lds_b128(
        (gv4i_p)gA0, (lv4i_p)&sA[0][dA0], 0, 0);
    __builtin_amdgcn_global_load_async_to_lds_b128(
        (gv4i_p)gA1, (lv4i_p)&sA[0][dA1], 0, 0);
    __builtin_amdgcn_global_load_async_to_lds_b128(
        (gv4i_p)gB,  (lv4i_p)&sB[0][dB], 0, 0);
    int cur = 0;
    for (int kk = 0; kk < K; kk += 32) {
        const bool more = (kk + 32) < K;
        wait_async0();          // buffer `cur` fully landed (our ops)
        __syncthreads();        // ... for every wave; prev compute done everywhere
        if (more) {             // DMA next tile while we do WMMAs on `cur`
            __builtin_amdgcn_global_load_async_to_lds_b128(
                (gv4i_p)(gA0 + kk + 32), (lv4i_p)&sA[cur ^ 1][dA0], 0, 0);
            __builtin_amdgcn_global_load_async_to_lds_b128(
                (gv4i_p)(gA1 + kk + 32), (lv4i_p)&sA[cur ^ 1][dA1], 0, 0);
            __builtin_amdgcn_global_load_async_to_lds_b128(
                (gv4i_p)(gB  + kk + 32), (lv4i_p)&sB[cur ^ 1][dB], 0, 0);
        }
#else
    // ---- register-staged double-buffer pipeline ----
    uint4 ra0 = *reinterpret_cast<const uint4*>(gA0);
    uint4 ra1 = *reinterpret_cast<const uint4*>(gA1);
    uint4 rb  = *reinterpret_cast<const uint4*>(gB);
    int cur = 0;
    for (int kk = 0; kk < K; kk += 32) {
        const bool more = (kk + 32) < K;
        *reinterpret_cast<uint4*>(&sA[cur][dA0]) = ra0;
        *reinterpret_cast<uint4*>(&sA[cur][dA1]) = ra1;
        *reinterpret_cast<uint4*>(&sB[cur][dB])  = rb;
        __syncthreads();
        if (more) {             // global loads overlap the WMMAs below
            ra0 = *reinterpret_cast<const uint4*>(gA0 + kk + 32);
            ra1 = *reinterpret_cast<const uint4*>(gA1 + kk + 32);
            rb  = *reinterpret_cast<const uint4*>(gB  + kk + 32);
        }
#endif
        // fragments from buffer `cur`
        union { struct { uint4 lo, hi; } u; v16bf v; } fa0, fa1, fb0, fb1;
        fa0.u.lo = *reinterpret_cast<const uint4*>(&sA[cur][(ma0 + lrow) * LDT + khalf]);
        fa0.u.hi = *reinterpret_cast<const uint4*>(&sA[cur][(ma0 + lrow) * LDT + 16 + khalf]);
        fa1.u.lo = *reinterpret_cast<const uint4*>(&sA[cur][(ma1 + lrow) * LDT + khalf]);
        fa1.u.hi = *reinterpret_cast<const uint4*>(&sA[cur][(ma1 + lrow) * LDT + 16 + khalf]);
        fb0.u.lo = *reinterpret_cast<const uint4*>(&sB[cur][(nb0 + lrow) * LDT + khalf]);
        fb0.u.hi = *reinterpret_cast<const uint4*>(&sB[cur][(nb0 + lrow) * LDT + 16 + khalf]);
        fb1.u.lo = *reinterpret_cast<const uint4*>(&sB[cur][(nb1 + lrow) * LDT + khalf]);
        fb1.u.hi = *reinterpret_cast<const uint4*>(&sB[cur][(nb1 + lrow) * LDT + 16 + khalf]);

        acc00 = __builtin_amdgcn_wmma_f32_16x16x32_bf16(
                    false, fa0.v, false, fb0.v, (short)0, acc00, false, false);
        acc01 = __builtin_amdgcn_wmma_f32_16x16x32_bf16(
                    false, fa0.v, false, fb1.v, (short)0, acc01, false, false);
        acc10 = __builtin_amdgcn_wmma_f32_16x16x32_bf16(
                    false, fa1.v, false, fb0.v, (short)0, acc10, false, false);
        acc11 = __builtin_amdgcn_wmma_f32_16x16x32_bf16(
                    false, fa1.v, false, fb1.v, (short)0, acc11, false, false);
        cur ^= 1;
    }

    // C/D layout: VGPR r -> M = r (lanes 0-15) or M = 8+r (lanes 16-31); N = lane&15
    union { v8f v; float f[8]; } c[4];
    c[0].v = acc00; c[1].v = acc01; c[2].v = acc10; c[3].v = acc11;
    const int hi8 = (lane >> 4) << 3;
    const int mA[2]  = { m_base + ma0 + hi8, m_base + ma1 + hi8 };
    const int nB[2]  = { n_base + nb0 + lrow, n_base + nb1 + lrow };
    const float bs[2] = { bias[nB[0]], bias[nB[1]] };

    #pragma unroll
    for (int a = 0; a < 2; ++a) {
        #pragma unroll
        for (int b = 0; b < 2; ++b) {
            const float* cf = c[a * 2 + b].f;
            #pragma unroll
            for (int r = 0; r < 8; ++r) {
                size_t idx = (size_t)(mA[a] + r) * N + nB[b];
                if (EPI == 0) {
                    out_or_v[idx] = cf[r] + bs[b];
                } else {
                    float I = cf[r] + bs[b];
                    float v = out_or_v[idx];
                    v += (I - v) * TAU_INV;
                    float s = (v > V_TH) ? 1.0f : 0.0f;
                    out_or_v[idx] = v * (1.0f - s);
                    ssum[idx] += s;
                }
            }
        }
    }
}

static inline int cdiv(int a, int b) { return (a + b - 1) / b; }

extern "C" void kernel_launch(void* const* d_in, const int* in_sizes, int n_in,
                              void* d_out, int out_size, void* d_ws, size_t ws_size,
                              hipStream_t stream) {
    const float* x    = (const float*)d_in[0];
    const float* W1   = (const float*)d_in[1];
    const float* b1   = (const float*)d_in[2];
    const float* W2   = (const float*)d_in[3];
    const float* b2   = (const float*)d_in[4];
    const float* Wout = (const float*)d_in[5];
    const float* bout = (const float*)d_in[6];

    const int Bn = 8192, Din = 256, H = 1024, Dout = 64, T = 15;

    char* p = (char*)d_ws;
    auto alloc = [&](size_t bytes) -> char* {
        char* r = p;
        p += (bytes + 255) & ~(size_t)255;
        return r;
    };
    unsigned short* xb    = (unsigned short*)alloc((size_t)Bn * Din * 2);
    unsigned short* W1b   = (unsigned short*)alloc((size_t)H * Din * 2);
    unsigned short* W2b   = (unsigned short*)alloc((size_t)H * H * 2);
    unsigned short* Woutb = (unsigned short*)alloc((size_t)Dout * H * 2);
    float* I1   = (float*)alloc((size_t)Bn * H * 4);
    float* v1   = (float*)alloc((size_t)Bn * H * 4);
    float* v2   = (float*)alloc((size_t)Bn * H * 4);
    float* sacc = (float*)alloc((size_t)Bn * H * 4);
    unsigned short* s1b = (unsigned short*)alloc((size_t)Bn * H * 2); // spikes, reused as r_bf16

    // --- weights / inputs to bf16 once ---
    cvt_bf16_kernel<<<cdiv(Bn * Din, 256), 256, 0, stream>>>(x,    xb,    1.0f, Bn * Din);
    cvt_bf16_kernel<<<cdiv(H * Din, 256),  256, 0, stream>>>(W1,   W1b,   1.0f, H * Din);
    cvt_bf16_kernel<<<cdiv(H * H, 256),    256, 0, stream>>>(W2,   W2b,   1.0f, H * H);
    cvt_bf16_kernel<<<cdiv(Dout * H, 256), 256, 0, stream>>>(Wout, Woutb, 1.0f, Dout * H);

    // --- zero state ---
    (void)hipMemsetAsync(v1,   0, (size_t)Bn * H * 4, stream);
    (void)hipMemsetAsync(v2,   0, (size_t)Bn * H * 4, stream);
    (void)hipMemsetAsync(sacc, 0, (size_t)Bn * H * 4, stream);

    // --- I1 = x @ W1^T + b1 (hoisted out of the time loop) ---
    {
        dim3 grid(H / BLK_N, Bn / BLK_M);
        wmma_gemm_kernel<0><<<grid, 256, 0, stream>>>(xb, W1b, b1, I1, nullptr, Bn, H, Din);
    }

    // --- T time steps: LIF1 (elementwise) -> GEMM2 with fused LIF2 ---
    {
        const int n4 = Bn * H / 4;
        dim3 grid2(H / BLK_N, Bn / BLK_M);
        for (int t = 0; t < T; ++t) {
            lif1_kernel<<<cdiv(n4, 256), 256, 0, stream>>>(
                (const float4*)I1, (float4*)v1, s1b, n4);
            wmma_gemm_kernel<1><<<grid2, 256, 0, stream>>>(
                s1b, W2b, b2, v2, sacc, Bn, H, H);
        }
    }

    // --- out = (ssum/15) @ Wout^T + bout ---
    cvt_bf16_kernel<<<cdiv(Bn * H, 256), 256, 0, stream>>>(sacc, s1b, 1.0f / (float)T, Bn * H);
    {
        dim3 grid(Dout / BLK_N, Bn / BLK_M);   // (1, 64)
        wmma_gemm_kernel<0><<<grid, 256, 0, stream>>>(
            s1b, Woutb, bout, (float*)d_out, nullptr, Bn, Dout, H);
    }
}
